// LSTM_88562225643594
// MI455X (gfx1250) — matrix operand order repaired and checked by reading, not compiled, not run on hardware
//
#include <hip/hip_runtime.h>
#include <hip/hip_bf16.h>

typedef __attribute__((ext_vector_type(16))) __bf16 v16bf;
typedef __attribute__((ext_vector_type(8)))  __bf16 v8bf;
typedef __attribute__((ext_vector_type(8)))  float  v8f;

#define T_STEPS 1024
#define BATCH   64
#define DIM     512
#define HID     512
#define GATES   2048      // 4*HID
#define KTOT    1024      // DIM + HID
#define NWG     32        // each owns 16 h-columns
#define NTHR    128       // 4 waves, one 16x16 M-tile each (M = 64)

#if __has_builtin(__builtin_amdgcn_global_load_async_to_lds_b128) && \
    __has_builtin(__builtin_amdgcn_s_wait_asynccnt)
#define HAVE_ASYNC_LDS 1
#else
#define HAVE_ASYNC_LDS 0
#endif

__device__ __forceinline__ __bf16 f2bf(float f) {
    union { float f; unsigned u; } v; v.f = f;
    unsigned r = v.u + 0x7FFFu + ((v.u >> 16) & 1u);   // round-to-nearest-even
    unsigned short h = (unsigned short)(r >> 16);
    return __builtin_bit_cast(__bf16, h);
}

// ---------------------------------------------------------------------------
// Prep 1: W = [Wi ; Wh] -> bf16, transposed to column-major [gatecol][K]
// ---------------------------------------------------------------------------
__global__ void lstm_prep_w(const float* __restrict__ Wi, const float* __restrict__ Wh,
                            __bf16* __restrict__ Wt) {
    int idx = blockIdx.x * blockDim.x + threadIdx.x;   // 0 .. GATES*KTOT-1
    int col = idx >> 10;
    int k   = idx & (KTOT - 1);
    float w = (k < DIM) ? Wi[(size_t)k * GATES + col]
                        : Wh[(size_t)(k - DIM) * GATES + col];
    Wt[idx] = f2bf(w);
}

// ---------------------------------------------------------------------------
// Prep 2: xs -> bf16 (one-time, removes all cvt VALU from the hot loop)
// ---------------------------------------------------------------------------
__global__ void lstm_prep_x(const float* __restrict__ xs, __bf16* __restrict__ xbf) {
    size_t i = ((size_t)blockIdx.x * blockDim.x + threadIdx.x) * 8;
    float4 a = *(const float4*)(xs + i);
    float4 b = *(const float4*)(xs + i + 4);
    v8bf o;
    o[0] = f2bf(a.x); o[1] = f2bf(a.y); o[2] = f2bf(a.z); o[3] = f2bf(a.w);
    o[4] = f2bf(b.x); o[5] = f2bf(b.y); o[6] = f2bf(b.z); o[7] = f2bf(b.w);
    *(v8bf*)(xbf + i) = o;
}

// ---------------------------------------------------------------------------
// Prep 3: bias = bi + bh, h0 -> bf16 double buffer, step flags init.
// ---------------------------------------------------------------------------
__global__ void lstm_prep_state(const float* __restrict__ h0, const float* __restrict__ bi,
                                const float* __restrict__ bh, __bf16* __restrict__ hbuf,
                                float* __restrict__ biasT, int* __restrict__ flags) {
    int idx = blockIdx.x * blockDim.x + threadIdx.x;
    if (idx < GATES)       biasT[idx] = bi[idx] + bh[idx];
    if (idx < BATCH * HID) hbuf[idx]  = f2bf(h0[idx]);
    if (idx <= T_STEPS)    flags[idx] = (idx == 0) ? NWG : 0;
}

// ---------------------------------------------------------------------------
// Persistent recurrent kernel (see round-1 analysis for the partitioning).
// ---------------------------------------------------------------------------
__global__ __launch_bounds__(NTHR, 1) void lstm_recurrent(
    const __bf16* __restrict__ xbf, const float* __restrict__ c0,
    const __bf16* __restrict__ Wt, const float* __restrict__ biasT,
    __bf16* __restrict__ hbuf, int* __restrict__ flags, float* __restrict__ out)
{
    extern __shared__ __bf16 wlds[];   // [4][16][KTOT] bf16 = 128 KB

    const int tid    = threadIdx.x;
    const int wg     = blockIdx.x;
    const int wave   = tid >> 5;
    const int lane   = tid & 31;
    const int laneLo = lane & 15;
    const int laneHi = lane >> 4;

    // ---- stage weight slice into LDS (4 contiguous 32 KB chunks) ----
#if HAVE_ASYNC_LDS
    {
        typedef int i32x4 __attribute__((vector_size(16)));
        typedef __attribute__((address_space(1))) i32x4* gp_t;   // global int4*
        typedef __attribute__((address_space(3))) i32x4* lp_t;   // LDS int4*
        for (int g = 0; g < 4; ++g) {
            const __bf16* src = Wt + (size_t)(g * HID + wg * 16) * KTOT;
            __bf16*       dst = wlds + (size_t)g * 16 * KTOT;
            for (int i = tid; i < (16 * KTOT) / 8; i += NTHR) {   // 16 B per lane per issue
                gp_t gs = (gp_t)(uintptr_t)(src + (size_t)i * 8);
                lp_t ld = (lp_t)(uintptr_t)(dst + (size_t)i * 8);
                __builtin_amdgcn_global_load_async_to_lds_b128(gs, ld, 0, 0);
            }
        }
        __builtin_amdgcn_s_wait_asynccnt(0);
    }
#else
    for (int g = 0; g < 4; ++g) {
        const uint4* src = (const uint4*)(Wt + (size_t)(g * HID + wg * 16) * KTOT);
        uint4*       dst = (uint4*)(wlds + (size_t)g * 16 * KTOT);
        for (int i = tid; i < (16 * KTOT) / 8; i += NTHR)
            dst[i] = src[i];
    }
#endif
    __syncthreads();

    const int mtile   = wave;
    const int arow    = mtile * 16 + laneLo;   // A-matrix row (16-bit A layout)
    const int ncol    = wg * 16 + laneLo;      // global column for B / C / D
    const int khalf8  = laneHi * 16;           // A per-lane K sub-offset
    const int khalf16 = laneHi * 16;           // B per-lane K sub-offset

    float bias_g[4];
    #pragma unroll
    for (int g = 0; g < 4; ++g) bias_g[g] = biasT[g * HID + ncol];

    // cell state tile in C/D accumulator layout: VGPR r -> M = 16*mtile + 8*laneHi + r
    v8f creg;
    #pragma unroll
    for (int r = 0; r < 8; ++r) {
        int M = mtile * 16 + laneHi * 8 + r;
        creg[r] = c0[(size_t)M * HID + ncol];
    }

    const size_t hsOff = (size_t)T_STEPS * BATCH * HID;
    const size_t cOff  = hsOff + (size_t)BATCH * HID;

    for (int t = 0; t < T_STEPS; ++t) {
        // ---- wait until h_t is fully published by all 32 workgroups ----
        while (__hip_atomic_load(&flags[t], __ATOMIC_ACQUIRE, __HIP_MEMORY_SCOPE_AGENT) < NWG)
            __builtin_amdgcn_s_sleep(1);
        __syncthreads();

        const __bf16* hsrc = hbuf + (size_t)(t & 1) * BATCH * HID;
        __bf16*       hdst = hbuf + (size_t)((t + 1) & 1) * BATCH * HID;

        v8f acc[4];
        #pragma unroll
        for (int g = 0; g < 4; ++g)
            #pragma unroll
            for (int r = 0; r < 8; ++r) acc[g][r] = bias_g[g];

        // ---- K = 0..511 : x_t contribution (bf16, pre-converted) ----
        const __bf16* xrow = xbf + ((size_t)t * BATCH + arow) * DIM;
        if (t + 1 < T_STEPS)
            __builtin_prefetch(xbf + ((size_t)(t + 1) * BATCH + arow) * DIM + khalf8, 0, 1);
        #pragma unroll 4
        for (int k = 0; k < 16; ++k) {
            const __bf16* p0 = xrow + k * 32 + khalf8;
            v8bf lo = *(const v8bf*)(p0);
            v8bf hi = *(const v8bf*)(p0 + 8);
            v16bf a = __builtin_shufflevector(lo, hi,
                        0,1,2,3,4,5,6,7,8,9,10,11,12,13,14,15);
            #pragma unroll
            for (int g = 0; g < 4; ++g) {
                const __bf16* wp = wlds + (size_t)(g * 16 + laneLo) * KTOT + k * 32 + khalf16;
                v16bf b = *(const v16bf*)wp;
                acc[g] = __builtin_amdgcn_wmma_f32_16x16x32_bf16(
                             false, a, false, b, (short)0, acc[g], false, false);
            }
        }

        // ---- K = 512..1023 : h_t contribution (bf16) ----
        const __bf16* hrow = hsrc + (size_t)arow * HID;
        #pragma unroll 4
        for (int k = 0; k < 16; ++k) {
            const __bf16* p0 = hrow + k * 32 + khalf8;
            v8bf lo = *(const v8bf*)(p0);
            v8bf hi = *(const v8bf*)(p0 + 8);
            v16bf a = __builtin_shufflevector(lo, hi,
                        0,1,2,3,4,5,6,7,8,9,10,11,12,13,14,15);
            #pragma unroll
            for (int g = 0; g < 4; ++g) {
                const __bf16* wp = wlds + (size_t)(g * 16 + laneLo) * KTOT
                                        + (DIM + k * 32) + khalf16;
                v16bf b = *(const v16bf*)wp;
                acc[g] = __builtin_amdgcn_wmma_f32_16x16x32_bf16(
                             false, a, false, b, (short)0, acc[g], false, false);
            }
        }

        // ---- elementwise LSTM cell + outputs ----
        #pragma unroll
        for (int r = 0; r < 8; ++r) {
            float xi = acc[0][r], xf = acc[1][r], xg = acc[2][r], xo = acc[3][r];
            float si = 1.f / (1.f + __expf(-xi));
            float sf = 1.f / (1.f + __expf(-xf));
            float so = 1.f / (1.f + __expf(-xo));
            float e2 = __expf(2.f * xg);
            float tg = (e2 - 1.f) / (e2 + 1.f);
            float cn = sf * creg[r] + si * tg;
            creg[r]  = cn;
            float hn = so * cn;
            int   M  = mtile * 16 + laneHi * 8 + r;
            out[((size_t)t * BATCH + M) * HID + ncol] = hn;       // hs[t]
            hdst[(size_t)M * HID + ncol] = f2bf(hn);
            if (t == T_STEPS - 1) {
                out[hsOff + (size_t)M * HID + ncol] = hn;         // final h
                out[cOff  + (size_t)M * HID + ncol] = cn;         // final c
            }
        }

        // ---- publish h_{t+1} ----
        __threadfence();
        __syncthreads();
        if (tid == 0)
            __hip_atomic_fetch_add(&flags[t + 1], 1, __ATOMIC_RELEASE,
                                   __HIP_MEMORY_SCOPE_AGENT);
    }
}

// ---------------------------------------------------------------------------
extern "C" void kernel_launch(void* const* d_in, const int* in_sizes, int n_in,
                              void* d_out, int out_size, void* d_ws, size_t ws_size,
                              hipStream_t stream) {
    (void)in_sizes; (void)n_in; (void)out_size; (void)ws_size;
    const float* xs = (const float*)d_in[0];
    const float* h0 = (const float*)d_in[1];
    const float* c0 = (const float*)d_in[2];
    const float* Wi = (const float*)d_in[3];
    const float* bi = (const float*)d_in[4];
    const float* Wh = (const float*)d_in[5];
    const float* bh = (const float*)d_in[6];
    float* out = (float*)d_out;

    // workspace layout (fixed byte offsets, ~72 MB total)
    char* ws = (char*)d_ws;
    __bf16* Wt    = (__bf16*)(ws);                       // 2048*1024 bf16 = 4 MB
    float*  biasT = (float*)(ws + (4u << 20));           // 2048 f32
    __bf16* hbuf  = (__bf16*)(ws + (4u << 20) + 8192);   // 2*64*512 bf16 double buffer
    int*    flags = (int*)(ws + (4u << 20) + 8192 + 131072);   // T_STEPS+1 ints
    __bf16* xbf   = (__bf16*)(ws + (8u << 20));          // 1024*64*512 bf16 = 64 MB

    lstm_prep_w<<<(GATES * KTOT) / 256, 256, 0, stream>>>(Wi, Wh, Wt);
    lstm_prep_x<<<(T_STEPS * BATCH * DIM) / (256 * 8), 256, 0, stream>>>(xs, xbf);
    lstm_prep_state<<<(BATCH * HID) / 256, 256, 0, stream>>>(h0, bi, bh, hbuf, biasT, flags);
    lstm_recurrent<<<NWG, NTHR, 4 * 16 * KTOT * (int)sizeof(__bf16), stream>>>(
        xbf, c0, Wt, biasT, hbuf, flags, out);
}
